// DHDN_25065429139907
// MI455X (gfx1250) — matrix operand order repaired and checked by reading, not compiled, not run on hardware
//
#include <hip/hip_runtime.h>
#include <hip/hip_bf16.h>
#include <math.h>

typedef _Float16 half_t;
typedef __attribute__((ext_vector_type(16))) _Float16 v16h_t;
typedef __attribute__((ext_vector_type(8)))  _Float16 v8h_t;
typedef __attribute__((ext_vector_type(8)))  float    v8f_t;

#define NP_CHUNK 19

static __device__ __forceinline__ int imin(int a, int b) { return a < b ? a : b; }
static __device__ __forceinline__ int imax(int a, int b) { return a > b ? a : b; }

// ---------------------------------------------------------------------------
// Weight prep: f32 -> f16, layout preserved ([co][K = ci*kh*kw])
// ---------------------------------------------------------------------------
__global__ void wprep_k(const float* __restrict__ w, half_t* __restrict__ o, int n) {
    int idx = blockIdx.x * 256 + threadIdx.x;
    if (idx < n) o[idx] = (half_t)w[idx];
}

// ---------------------------------------------------------------------------
// Bilinear crop-resize: one batch image -> 19 patches [j][3][224][224] f16
// ---------------------------------------------------------------------------
__global__ void crop_resize_k(const float* __restrict__ img, const float* __restrict__ uv,
                              half_t* __restrict__ patches) {
    int j = blockIdx.x, oy = blockIdx.y, ox = threadIdx.x;
    float px = uv[j * 2 + 0], py = uv[j * 2 + 1];
    int cx = (int)floorf((px + 0.5f) * 512.f);
    int cy = (int)floorf((py + 0.5f) * 512.f);
    int x_min = imax(0, cx - 16), x_max = imin(512, cx + 16);
    if (x_max <= x_min) x_min = x_max - 1;
    int y_min = imax(0, cy - 16), y_max = imin(512, cy + 16);
    if (y_max <= y_min) y_min = y_max - 1;

    float ny = (float)(y_max - y_min);
    float sy = fmaxf(0.f, ((float)oy + 0.5f) * ny / 224.f - 0.5f);
    int iy0 = (int)floorf(sy);
    float wy = sy - (float)iy0;
    int iy1 = imin(iy0 + 1, y_max - y_min - 1);
    int y0 = y_min + iy0, y1 = y_min + iy1;

    float nx = (float)(x_max - x_min);
    float sx = fmaxf(0.f, ((float)ox + 0.5f) * nx / 224.f - 0.5f);
    int ix0 = (int)floorf(sx);
    float wx = sx - (float)ix0;
    int ix1 = imin(ix0 + 1, x_max - x_min - 1);
    int x0 = x_min + ix0, x1 = x_min + ix1;

    for (int c = 0; c < 3; ++c) {
        const float* imc = img + (size_t)c * 512 * 512;
        float top = imc[y0 * 512 + x0] * (1.f - wx) + imc[y0 * 512 + x1] * wx;
        float bot = imc[y1 * 512 + x0] * (1.f - wx) + imc[y1 * 512 + x1] * wx;
        float v = top * (1.f - wy) + bot * wy;
        patches[(((size_t)j * 3 + c) * 224 + oy) * 224 + ox] = (half_t)v;
    }
}

// ---------------------------------------------------------------------------
// Implicit-GEMM conv with v_wmma_f32_16x16x32_f16.
//   A[m,k] = im2col(input) staged [m][k] in LDS
//   B[k,n] = weights, staged [n][k] in LDS (weights kept in natural [co][K])
// Tile 64(M) x 64(N), K-step 32. 256 threads = 8 wave32; waves in 4(M) x 2(N).
// Fragment k order is contiguous, k = 16*h + s, in both A and B; all fragment traffic is ds_load_b128.
// Row stride 40 halves (80B): 16B-aligned vectors, conflict-free bank stride.
// Epilogue (vectorized 16B): f = acc*gam[n] + bet[n] (+ residual) (+ relu) -> f16.
// ---------------------------------------------------------------------------
__global__ void conv_wmma_k(const half_t* __restrict__ in, const half_t* __restrict__ w16,
                            const float* __restrict__ gam, const float* __restrict__ bet,
                            const half_t* __restrict__ res, half_t* __restrict__ out,
                            int ci, int hin, int win, int co, int ho, int wo,
                            int stride_, int kh, int kw, int pad, int do_relu) {
    const int STR = 40;
    __shared__ __align__(16) half_t As[64 * STR];
    __shared__ __align__(16) half_t Bs[64 * STR];

    const int M = ho * wo;
    const int khkw = kh * kw;
    const int Ktot = ci * khkw;
    const int tile_m = blockIdx.x * 64;
    const int tile_n = blockIdx.y * 64;

    const size_t z = blockIdx.z;
    in += z * (size_t)ci * hin * win;
    out += z * (size_t)co * M;
    if (res) res += z * (size_t)co * M;

    const int tid = threadIdx.x;
    const int lane = tid & 31;
    const int wv = tid >> 5;
    const int wm = wv & 3;      // 0..3 : 16-row M subtile
    const int wn = wv >> 2;     // 0..1 : 32-col N subtile

    v8f_t acc0 = {};
    v8f_t acc1 = {};

    const int l16 = (lane & 16) ? 1 : 0;
    const int ncol = lane & 15;
    const int mrow = wm * 16 + ncol;
    const int rowB0 = wn * 32 + ncol;
    const int rowB1 = rowB0 + 16;
    const int kbase = l16 * 16;

    // B staging coords (one 16B vector per thread per K-step)
    const int browS = tid & 63;
    const int bsegS = (tid >> 6) * 8;

    for (int k0 = 0; k0 < Ktot; k0 += 32) {
        // ---- stage A tile (64 x 32) via im2col (gather; scalar) ----
        for (int e = 0; e < 8; ++e) {
            int idx = tid + e * 256;
            int m = idx >> 5, kk = idx & 31;
            half_t v = (half_t)0.f;
            int mg = tile_m + m;
            int kg = k0 + kk;
            if (mg < M && kg < Ktot) {
                int oh = mg / wo, ow = mg % wo;
                int c  = kg / khkw, rem = kg % khkw;
                int r = rem / kw, s = rem % kw;
                int ih = oh * stride_ - pad + r;
                int iw = ow * stride_ - pad + s;
                if (ih >= 0 && ih < hin && iw >= 0 && iw < win)
                    v = in[((size_t)c * hin + ih) * win + iw];
            }
            As[m * STR + kk] = v;
        }
        // ---- stage B tile, transposed to [n][k]; contiguous global reads ----
        {
            int ng = tile_n + browS;
            int kg0 = k0 + bsegS;
            v8h_t vb = {};
            if (ng < co) {
                const half_t* src = w16 + (size_t)ng * Ktot + kg0;
                if (kg0 + 7 < Ktot) {
                    __builtin_memcpy(&vb, src, sizeof(vb));
                } else {
#pragma unroll
                    for (int u = 0; u < 8; ++u)
                        if (kg0 + u < Ktot) vb[u] = src[u];
                }
            }
            *(v8h_t*)(Bs + browS * STR + bsegS) = vb;
        }
        __syncthreads();

        // ---- fragments: contiguous 16B LDS vector loads ----
        v8h_t a_lo = *(const v8h_t*)(As + mrow * STR + kbase);
        v8h_t a_hi = *(const v8h_t*)(As + mrow * STR + kbase + 8);
        v16h_t a = __builtin_shufflevector(a_lo, a_hi,
                                           0, 1, 2, 3, 4, 5, 6, 7,
                                           8, 9, 10, 11, 12, 13, 14, 15);
        v8h_t b0l = *(const v8h_t*)(Bs + rowB0 * STR + kbase);
        v8h_t b0h = *(const v8h_t*)(Bs + rowB0 * STR + kbase + 8);
        v16h_t b0 = __builtin_shufflevector(b0l, b0h,
                                            0, 1, 2, 3, 4, 5, 6, 7,
                                            8, 9, 10, 11, 12, 13, 14, 15);
        v8h_t b1l = *(const v8h_t*)(Bs + rowB1 * STR + kbase);
        v8h_t b1h = *(const v8h_t*)(Bs + rowB1 * STR + kbase + 8);
        v16h_t b1 = __builtin_shufflevector(b1l, b1h,
                                            0, 1, 2, 3, 4, 5, 6, 7,
                                            8, 9, 10, 11, 12, 13, 14, 15);

        acc0 = __builtin_amdgcn_wmma_f32_16x16x32_f16(false, a, false, b0, (short)0, acc0, false, false);
        acc1 = __builtin_amdgcn_wmma_f32_16x16x32_f16(false, a, false, b1, (short)0, acc1, false, false);
        __syncthreads();
    }

    // ---- epilogue: per lane, 8 accumulator rows are contiguous in m ----
    const int mbase = tile_m + wm * 16 + l16 * 8;
#pragma unroll
    for (int sub = 0; sub < 2; ++sub) {
        int ng = tile_n + wn * 32 + sub * 16 + ncol;
        if (ng >= co) continue;
        const v8f_t& acc = sub ? acc1 : acc0;
        float ga = gam[ng], be = bet[ng];
        half_t* op = out + (size_t)ng * M + mbase;
        if (mbase + 7 < M) {
            v8h_t rv = {};
            if (res) __builtin_memcpy(&rv, res + (size_t)ng * M + mbase, sizeof(rv));
            v8h_t ov;
#pragma unroll
            for (int r = 0; r < 8; ++r) {
                float f = acc[r] * ga + be;
                if (res) f += (float)rv[r];
                if (do_relu) f = fmaxf(f, 0.f);
                ov[r] = (half_t)f;
            }
            __builtin_memcpy(op, &ov, sizeof(ov));
        } else {
#pragma unroll
            for (int r = 0; r < 8; ++r) {
                if (mbase + r >= M) break;
                float f = acc[r] * ga + be;
                if (res) f += (float)res[(size_t)ng * M + mbase + r];
                if (do_relu) f = fmaxf(f, 0.f);
                op[r] = (half_t)f;
            }
        }
    }
}

// ---------------------------------------------------------------------------
// 3x3 stride-2 pad-1 maxpool: [19][64][112][112] -> [19][64][56][56], f16
// ---------------------------------------------------------------------------
__global__ void maxpool_k(const half_t* __restrict__ in, half_t* __restrict__ out) {
    int idx = blockIdx.x * 256 + threadIdx.x;
    const int total = NP_CHUNK * 64 * 56 * 56;
    if (idx >= total) return;
    int x = idx % 56;
    int y = (idx / 56) % 56;
    int pc = idx / (56 * 56);  // p*64 + c
    const half_t* ip = in + (size_t)pc * 112 * 112;
    float mx = -3.4e38f;
    for (int dy = 0; dy < 3; ++dy) {
        int iy = y * 2 - 1 + dy;
        if (iy < 0 || iy >= 112) continue;
        for (int dx = 0; dx < 3; ++dx) {
            int ix = x * 2 - 1 + dx;
            if (ix < 0 || ix >= 112) continue;
            mx = fmaxf(mx, (float)ip[iy * 112 + ix]);
        }
    }
    out[idx] = (half_t)mx;
}

// ---------------------------------------------------------------------------
// Global average pool over 7x7: [19][512][49] f16 -> pool[(base+j)*512+c] f32
// ---------------------------------------------------------------------------
__global__ void avgpool_k(const half_t* __restrict__ in, float* __restrict__ pool, int base) {
    int idx = blockIdx.x * 256 + threadIdx.x;
    if (idx >= NP_CHUNK * 512) return;
    int j = idx / 512, c = idx % 512;
    const half_t* ip = in + (size_t)(j * 512 + c) * 49;
    float s = 0.f;
    for (int k = 0; k < 49; ++k) s += (float)ip[k];
    pool[(size_t)(base + j) * 512 + c] = s * (1.f / 49.f);
}

// ---------------------------------------------------------------------------
// Generic dense layer: y[row,o] = [relu]( x[row,:] . w[o,:] + bias[o] )
// ---------------------------------------------------------------------------
__global__ void linear_k(const float* __restrict__ x, const float* __restrict__ w,
                         const float* __restrict__ bias, float* __restrict__ y,
                         int K, int O, int do_relu) {
    int row = blockIdx.x;
    int o = blockIdx.y * 256 + threadIdx.x;
    if (o >= O) return;
    const float* xr = x + (size_t)row * K;
    const float* wr = w + (size_t)o * K;
    float acc = bias ? bias[o] : 0.f;
    for (int k = 0; k < K; ++k) acc += xr[k] * wr[k];
    if (do_relu) acc = fmaxf(acc, 0.f);
    y[(size_t)row * O + o] = acc;
}

// ---------------------------------------------------------------------------
// dist = 0.5*pdist(points) + 0.5*pdist(feats); 4 nearest (incl self), tie -> low idx
// grid (B, J), 32 threads
// ---------------------------------------------------------------------------
__global__ void topk_k(const float* __restrict__ uv, const float* __restrict__ feats,
                       int* __restrict__ idxout) {
    int b = blockIdx.x, s = blockIdx.y;
    int t = threadIdx.x;
    __shared__ float drow[19];
    if (t < 19) {
        float dx = uv[b * 38 + s * 2] - uv[b * 38 + t * 2];
        float dy = uv[b * 38 + s * 2 + 1] - uv[b * 38 + t * 2 + 1];
        float dp = sqrtf(fmaxf(dx * dx + dy * dy, 0.f));
        const float* fs = feats + ((size_t)b * 19 + s) * 256;
        const float* ft = feats + ((size_t)b * 19 + t) * 256;
        float df2 = 0.f;
        for (int e = 0; e < 256; ++e) { float d = fs[e] - ft[e]; df2 += d * d; }
        drow[t] = 0.5f * dp + 0.5f * sqrtf(fmaxf(df2, 0.f));
    }
    __syncthreads();
    if (t == 0) {
        bool used[19];
        for (int u = 0; u < 19; ++u) used[u] = false;
        for (int kk = 0; kk < 4; ++kk) {
            int best = -1; float bv = 3.4e38f;
            for (int u = 0; u < 19; ++u)
                if (!used[u] && drow[u] < bv) { bv = drow[u]; best = u; }
            used[best] = true;
            idxout[((size_t)b * 19 + s) * 4 + kk] = best;
        }
    }
}

// ---------------------------------------------------------------------------
// Hypergraph conv: segment-sum message passing. block per batch, 256 thr = E
// ---------------------------------------------------------------------------
__global__ void hconv_k(const float* __restrict__ xw, const int* __restrict__ idx,
                        const float* __restrict__ hgb, float* __restrict__ g) {
    int b = blockIdx.x;
    int e = threadIdx.x;
    __shared__ float m_s[19 * 256];
    __shared__ float binv[19];
    __shared__ int row1[76];
    if (e < 76) row1[e] = idx[b * 76 + e];
#pragma unroll
    for (int j = 0; j < 19; ++j) m_s[j * 256 + e] = 0.f;
    __syncthreads();
    if (e < 19) {
        int c = 0;
        for (int i = 0; i < 76; ++i) c += (row1[i] == e) ? 1 : 0;
        binv[e] = (c > 0) ? (1.f / (float)c) : 0.f;
    }
    for (int i = 0; i < 76; ++i)
        m_s[row1[i] * 256 + e] += xw[((size_t)b * 19 + (i >> 2)) * 256 + e];
    __syncthreads();
    float hb = hgb[e];
    for (int j = 0; j < 19; ++j) {
        float acc = 0.f;
        for (int kk = 0; kk < 4; ++kk) {
            int r = row1[j * 4 + kk];
            acc += m_s[r * 256 + e] * binv[r];
        }
        g[((size_t)b * 19 + j) * 256 + e] = acc * 0.25f + hb;
    }
}

// ---------------------------------------------------------------------------
// Attention over the batch axis (per reference einsum): grid (J, NH), 256 thr
// thread = s*64 + d   (s: batch 0..3, d: head dim 0..63)
// ---------------------------------------------------------------------------
__global__ void attn_k(const float* __restrict__ qkv, float* __restrict__ ao) {
    int j = blockIdx.x, h = blockIdx.y;
    int tid = threadIdx.x;
    int s = tid >> 6, d = tid & 63;
    __shared__ float red[256];
    __shared__ float att[4][4];
    float q = qkv[((size_t)(s * 19 + j)) * 768 + h * 64 + d];
    for (int t = 0; t < 4; ++t) {
        red[tid] = q * qkv[((size_t)(t * 19 + j)) * 768 + 256 + h * 64 + d];
        __syncthreads();
        if (d == 0) {
            float sum = 0.f;
            for (int i = 0; i < 64; ++i) sum += red[s * 64 + i];
            att[s][t] = sum * 0.125f;  // 1/sqrt(64)
        }
        __syncthreads();
    }
    if (tid < 4) {
        int ss = tid;
        float mx = att[ss][0];
        for (int t = 1; t < 4; ++t) mx = fmaxf(mx, att[ss][t]);
        float ex[4], se = 0.f;
        for (int t = 0; t < 4; ++t) { ex[t] = expf(att[ss][t] - mx); se += ex[t]; }
        for (int t = 0; t < 4; ++t) att[ss][t] = ex[t] / se;
    }
    __syncthreads();
    float o = 0.f;
    for (int t = 0; t < 4; ++t)
        o += att[s][t] * qkv[((size_t)(t * 19 + j)) * 768 + 512 + h * 64 + d];
    ao[((size_t)(s * 19 + j)) * 256 + h * 64 + d] = o;
}

// ---------------------------------------------------------------------------
// VAE reparameterization; writes mu / log_var / z into d_out + z to scratch
// ---------------------------------------------------------------------------
__global__ void reparam_k(const float* __restrict__ zp, const float* __restrict__ eps,
                          float* __restrict__ out, float* __restrict__ zbuf) {
    int i = threadIdx.x;  // 256 = 4*64
    int b = i >> 6, p = i & 63;
    float mu = zp[b * 128 + p];
    float lv = zp[b * 128 + 64 + p];
    float z = mu + eps[b * 64 + p] * expf(0.5f * lv);
    out[152 + b * 64 + p] = mu;
    out[408 + b * 64 + p] = lv;
    out[664 + b * 64 + p] = z;
    zbuf[b * 64 + p] = z;
}

// ---------------------------------------------------------------------------
// Host orchestration
// ---------------------------------------------------------------------------
extern "C" void kernel_launch(void* const* d_in, const int* in_sizes, int n_in,
                              void* d_out, int out_size, void* d_ws, size_t ws_size,
                              hipStream_t stream) {
    (void)in_sizes; (void)n_in; (void)out_size; (void)ws_size;
    const float* img = (const float*)d_in[0];
    const float* uv  = (const float*)d_in[1];
    const float* eps = (const float*)d_in[2];
    float* out = (float*)d_out;
    char* ws = (char*)d_ws;

    size_t off = 0;
    auto alloc = [&](size_t bytes) -> void* {
        off = (off + 255) & ~(size_t)255;
        void* p = (void*)(ws + off);
        off += bytes;
        return p;
    };

    // conv weight specs: {d_in index of w, K=ci*kh*kw, co}
    struct WSpec { int widx; int K; int co; };
    static const WSpec wspecs[20] = {
        {3, 147, 64},                                   // conv1 7x7
        {6, 576, 64},   {9, 576, 64},                   // b0 w1 w2
        {12, 576, 64},  {15, 576, 64},                  // b1
        {18, 576, 128}, {21, 1152, 128}, {24, 64, 128}, // b2 w1 w2 wd
        {27, 1152, 128},{30, 1152, 128},                // b3
        {33, 1152, 256},{36, 2304, 256}, {39, 128, 256},// b4
        {42, 2304, 256},{45, 2304, 256},                // b5
        {48, 2304, 512},{51, 4608, 512}, {54, 256, 512},// b6
        {57, 4608, 512},{60, 4608, 512},                // b7
    };
    half_t* wptr[20];
    for (int i = 0; i < 20; ++i)
        wptr[i] = (half_t*)alloc((size_t)wspecs[i].K * wspecs[i].co * sizeof(half_t));

    half_t* patch = (half_t*)alloc((size_t)2860032 * sizeof(half_t));   // 19*3*224*224
    half_t* bufA  = (half_t*)alloc((size_t)15253504 * sizeof(half_t));  // 19*64*112*112
    half_t* bufB  = (half_t*)alloc((size_t)3813376 * sizeof(half_t));   // 19*64*56*56
    half_t* bufC  = (half_t*)alloc((size_t)1906688 * sizeof(half_t));   // 19*128*28*28
    float* pool   = (float*)alloc((size_t)76 * 512 * sizeof(float));
    float* feats  = (float*)alloc((size_t)76 * 256 * sizeof(float));
    float* hbuf   = (float*)alloc((size_t)76 * 256 * sizeof(float));
    float* xwbuf  = (float*)alloc((size_t)76 * 256 * sizeof(float));
    float* gbuf   = (float*)alloc((size_t)76 * 256 * sizeof(float));
    float* qkvbuf = (float*)alloc((size_t)76 * 768 * sizeof(float));
    float* aobuf  = (float*)alloc((size_t)76 * 256 * sizeof(float));
    float* h2buf  = (float*)alloc((size_t)76 * 256 * sizeof(float));
    int*   idxbuf = (int*)alloc((size_t)304 * sizeof(int));
    float* z1buf  = (float*)alloc((size_t)4 * 256 * sizeof(float));
    float* zpbuf  = (float*)alloc((size_t)4 * 128 * sizeof(float));
    float* zbuf   = (float*)alloc((size_t)4 * 64 * sizeof(float));
    float* hrbuf  = (float*)alloc((size_t)4 * 4864 * sizeof(float));
    float* r1buf  = (float*)alloc((size_t)4 * 256 * sizeof(float));

    // 1) weight f32 -> f16 convert (once per launch; deterministic)
    for (int i = 0; i < 20; ++i) {
        int n = wspecs[i].K * wspecs[i].co;
        wprep_k<<<(n + 255) / 256, 256, 0, stream>>>(
            (const float*)d_in[wspecs[i].widx], wptr[i], n);
    }

    auto conv = [&](const half_t* in, int wi, int dpi, const half_t* res, half_t* o,
                    int ci, int hin, int co, int ho, int kh, int st, int pad, int relu) {
        dim3 g((unsigned)((ho * ho + 63) / 64), (unsigned)((co + 63) / 64), NP_CHUNK);
        conv_wmma_k<<<g, 256, 0, stream>>>(in, wptr[wi],
                                           (const float*)d_in[dpi + 1], (const float*)d_in[dpi + 2],
                                           res, o, ci, hin, hin, co, ho, ho, st, kh, kh, pad, relu);
    };

    struct Blk { int wi1, d1, wi2, d2, wiw, dw, ci, co, st; };
    static const Blk blks[8] = {
        {1, 6, 2, 9, -1, -1, 64, 64, 1},
        {3, 12, 4, 15, -1, -1, 64, 64, 1},
        {5, 18, 6, 21, 7, 24, 64, 128, 2},
        {8, 27, 9, 30, -1, -1, 128, 128, 1},
        {10, 33, 11, 36, 12, 39, 128, 256, 2},
        {13, 42, 14, 45, -1, -1, 256, 256, 1},
        {15, 48, 16, 51, 17, 54, 256, 512, 2},
        {18, 57, 19, 60, -1, -1, 512, 512, 1},
    };

    // 2) CNN, one batch image (19 patches) per chunk
    for (int b = 0; b < 4; ++b) {
        crop_resize_k<<<dim3(19, 224), 224, 0, stream>>>(
            img + (size_t)b * 3 * 512 * 512, uv + b * 38, patch);
        // conv1 7x7 s2 p3 + bn + relu
        conv(patch, 0, 3, nullptr, bufA, 3, 224, 64, 112, 7, 2, 3, 1);
        {
            int total = NP_CHUNK * 64 * 56 * 56;
            maxpool_k<<<(total + 255) / 256, 256, 0, stream>>>(bufA, bufB);
        }
        half_t* cur = bufB;
        int hsz = 56;
        for (int i = 0; i < 8; ++i) {
            const Blk& bl = blks[i];
            int ho = (bl.st == 2) ? hsz / 2 : hsz;
            conv(cur, bl.wi1, bl.d1, nullptr, bufA, bl.ci, hsz, bl.co, ho, 3, bl.st, 1, 1);
            const half_t* res = cur;  // identity shortcut
            if (bl.wiw >= 0) {
                conv(cur, bl.wiw, bl.dw, nullptr, bufC, bl.ci, hsz, bl.co, ho, 1, bl.st, 0, 0);
                res = bufC;
            }
            conv(bufA, bl.wi2, bl.d2, res, cur, bl.co, ho, bl.co, ho, 3, 1, 1, 1);
            hsz = ho;
        }
        avgpool_k<<<(NP_CHUNK * 512 + 255) / 256, 256, 0, stream>>>(cur, pool, b * 19);
    }

    // 3) head (tiny, f32)
    linear_k<<<dim3(76, 1), 256, 0, stream>>>(pool, (const float*)d_in[63], (const float*)d_in[64], feats, 512, 256, 0);
    linear_k<<<dim3(76, 1), 256, 0, stream>>>(uv, (const float*)d_in[65], (const float*)d_in[66], hbuf, 2, 256, 0);
    topk_k<<<dim3(4, 19), 32, 0, stream>>>(uv, feats, idxbuf);
    linear_k<<<dim3(76, 1), 256, 0, stream>>>(hbuf, (const float*)d_in[67], nullptr, xwbuf, 256, 256, 0);
    hconv_k<<<4, 256, 0, stream>>>(xwbuf, idxbuf, (const float*)d_in[68], gbuf);
    linear_k<<<dim3(76, 3), 256, 0, stream>>>(gbuf, (const float*)d_in[69], (const float*)d_in[70], qkvbuf, 256, 768, 0);
    attn_k<<<dim3(19, 4), 256, 0, stream>>>(qkvbuf, aobuf);
    linear_k<<<dim3(76, 1), 256, 0, stream>>>(aobuf, (const float*)d_in[71], (const float*)d_in[72], h2buf, 256, 256, 0);
    linear_k<<<dim3(4, 1), 256, 0, stream>>>(h2buf, (const float*)d_in[73], (const float*)d_in[74], z1buf, 4864, 256, 1);
    linear_k<<<dim3(4, 1), 256, 0, stream>>>(z1buf, (const float*)d_in[75], (const float*)d_in[76], zpbuf, 256, 128, 0);
    reparam_k<<<1, 256, 0, stream>>>(zpbuf, eps, out, zbuf);
    linear_k<<<dim3(4, 19), 256, 0, stream>>>(zbuf, (const float*)d_in[77], (const float*)d_in[78], hrbuf, 64, 4864, 0);
    linear_k<<<dim3(4, 1), 256, 0, stream>>>(hrbuf, (const float*)d_in[79], (const float*)d_in[80], r1buf, 4864, 256, 1);
    linear_k<<<dim3(4, 1), 256, 0, stream>>>(r1buf, (const float*)d_in[81], (const float*)d_in[82], out, 256, 38, 0);
}
